// AttentionHead_41875931136747
// MI455X (gfx1250) — matrix operand order repaired and checked by reading, not compiled, not run on hardware
//
#include <hip/hip_runtime.h>

// Fused attention head for MI455X (gfx1250, wave32, WMMA, TDM/async-LDS).
//   B=8, SQ=SK=2048, D_MODEL=1024, OUT_DIM=128
// K0: convert+transpose weights f32[1024][128] -> f16 Wt[128][1024] (x3)
// K1: QKV projection GEMM. Whole Wt (256KB) staged into LDS once per block
//     via TENSOR_LOAD_TO_LDS (TDM) or global_load_async_to_lds_b128 fallback.
//     Inner loop is 2x-unrolled ping-pong double buffering (no register
//     copies) with sched_group_barrier interleaving next-tile ds_loads
//     between current-tile WMMAs (partial dscnt waits).
// K2: per (batch, 16-query tile): scores -> LDS, masked 2-pass softmax,
//     P(f16) -> LDS, P*V via WMMA (ping-pong pipelined) -> f32 out.

typedef __attribute__((ext_vector_type(16))) _Float16 v16h;
typedef __attribute__((ext_vector_type(8)))  _Float16 v8h;
typedef __attribute__((ext_vector_type(8)))  float    v8f;
typedef __attribute__((ext_vector_type(4)))  float    v4f;
typedef unsigned int u32x4 __attribute__((ext_vector_type(4)));
typedef int          i32x4 __attribute__((ext_vector_type(4)));
typedef int          i32x8 __attribute__((ext_vector_type(8)));

#define B_   8
#define SQ_  2048
#define SK_  2048
#define DM_  1024
#define DO_  128
#define MASKVAL (-1.0e6f)

// LDS row strides chosen so stride % (64 banks * 4B) == 16B -> lanes r=0..15
// of a fragment read hit banks 4r..4r+3 (conflict-free).
#define WROWB 2064   // bytes per weight row in LDS (2048 data + 16 pad)
#define SROW  2052   // f32 elements per score row in LDS
#define PROW  2056   // f16 elements per P row in LDS

#if __has_builtin(__builtin_amdgcn_sched_group_barrier)
#define SGB(mask, size) __builtin_amdgcn_sched_group_barrier((mask), (size), 0)
#else
#define SGB(mask, size)
#endif
// mask bits: 0x008 MFMA/WMMA, 0x020 VMEM read, 0x100 DS read

// ---- fragment loaders (contiguous k order for A and B, wave32) ------------------------
// A (16x32 f16): lane r=lane&15 holds row M=r, hi=lane>>4;
//   halves 0..15 = K(k0+hi*16+0..15) (contiguous, matches B)
// B (32x16 f16): lane r holds col N; halves 0..15 = 16 consecutive K from
//   k0+hi*16 -> contiguous when B is stored N-major.

static __device__ __forceinline__ v16h load_a_from_f32(const float* base) {
  v4f x0 = *(const v4f*)(base + 0);
  v4f x1 = *(const v4f*)(base + 4);
  v4f x2 = *(const v4f*)(base + 8);
  v4f x3 = *(const v4f*)(base + 12);
  v16h a;
#pragma unroll
  for (int i = 0; i < 4; ++i) {
    a[i]      = (_Float16)x0[i];
    a[4 + i]  = (_Float16)x1[i];
    a[8 + i]  = (_Float16)x2[i];
    a[12 + i] = (_Float16)x3[i];
  }
  return a;
}

static __device__ __forceinline__ v16h load_a_from_f16(const _Float16* base) {
  v8h lo = *(const v8h*)(base + 0);
  v8h hh = *(const v8h*)(base + 8);
  v16h a;
#pragma unroll
  for (int i = 0; i < 8; ++i) { a[i] = lo[i]; a[8 + i] = hh[i]; }
  return a;
}

static __device__ __forceinline__ v16h load_b_contig(const _Float16* p) {
  v8h lo = *(const v8h*)(p + 0);
  v8h hh = *(const v8h*)(p + 8);
  v16h b;
#pragma unroll
  for (int i = 0; i < 8; ++i) { b[i] = lo[i]; b[8 + i] = hh[i]; }
  return b;
}

// Weight element (row, k) in the padded LDS image. kuni is the wave-uniform
// k base (multiple of 32); the 8B pad after every 1024B depends only on kuni
// ((kuni + hi*16) never crosses a 512-half boundary), so it folds to SALU.
static __device__ __forceinline__ const _Float16* lds_w_ptr(
    const unsigned char* lw, int row, int kuni, int hi) {
  const unsigned pad = (kuni >= 512) ? 8u : 0u;  // scalar select
  return (const _Float16*)(lw + (unsigned)row * WROWB +
                           (unsigned)(kuni + hi * 16) * 2u + pad);
}

// ---------------------------------------------------------------------------
// K0: weight convert+transpose: w[1024][128] f32 -> Wt[128][1024] f16 (x3)
// ---------------------------------------------------------------------------
__global__ void wconv_kernel(const float* __restrict__ wq,
                             const float* __restrict__ wk,
                             const float* __restrict__ wv,
                             _Float16* __restrict__ wt) {
  int idx = blockIdx.x * blockDim.x + threadIdx.x;
  const int per = DM_ * DO_;
  if (idx >= 3 * per) return;
  int sel = idx / per;
  int rem = idx - sel * per;
  int kk  = rem / DO_;
  int nn  = rem % DO_;
  const float* w = (sel == 0) ? wq : (sel == 1) ? wk : wv;
  wt[(size_t)sel * per + (size_t)nn * DM_ + kk] = (_Float16)w[(size_t)kk * DO_ + nn];
}

// ---------------------------------------------------------------------------
// K1: QKV projection. Block = 256 thr (8 waves). blockIdx/128 selects Q/K/V.
//     Whole Wt[sel] staged to LDS (padded, 264192 B) via TDM or async copies.
// ---------------------------------------------------------------------------
__global__ __launch_bounds__(256)
void qkv_proj_kernel(const float* __restrict__ q_in,
                     const float* __restrict__ k_in,
                     const float* __restrict__ v_in,
                     const float* __restrict__ b_q,
                     const float* __restrict__ b_k,
                     const float* __restrict__ b_v,
                     const _Float16* __restrict__ wt,
                     _Float16* __restrict__ qf,
                     _Float16* __restrict__ kf,
                     _Float16* __restrict__ vt) {
  extern __shared__ unsigned char smem[];
  unsigned char* lw = smem;  // padded weight image, 128 rows x WROWB bytes

  const int tid  = threadIdx.x;
  const int lane = tid & 31;
  const int r    = lane & 15;
  const int hi   = lane >> 4;
  const int sel  = blockIdx.x >> 7;                       // 0=Q 1=K 2=V
  const int m0   = (((blockIdx.x & 127) << 3) | (tid >> 5)) * 16;

  const float*    X   = (sel == 0) ? q_in : (sel == 1) ? k_in : v_in;
  const float*    bia = (sel == 0) ? b_q  : (sel == 1) ? b_k  : b_v;
  const _Float16* W   = wt + (size_t)sel * DO_ * DM_;

  // ---- stage Wt[sel] (128 x 1024 f16) into LDS ----------------------------
#if __has_builtin(__builtin_amdgcn_tensor_load_to_lds)
  if (tid < 32) {  // one wave issues the TDM descriptor
    unsigned long long ga = (unsigned long long)(uintptr_t)W;
    u32x4 g0;
    g0[0] = 1u;                                     // count=1, gather off
    g0[1] = (unsigned)(uintptr_t)lw;                // lds_addr (bytes)
    g0[2] = (unsigned)(ga & 0xFFFFFFFFu);           // global_addr[31:0]
    g0[3] = (unsigned)((ga >> 32) & 0x1FFFFFFu) | (2u << 30);  // addr[56:32]|type=2
    i32x8 g1;
    g1[0] = (int)((1u << 16) |                      // data_size = 2B
                  (1u << 20) |                      // pad_enable
                  (7u << 22) |                      // pad_interval = 256 dwords
                  (1u << 25));                      // pad_amount  = 2 dwords
    g1[1] = (int)(1024u << 16);                     // tensor_dim0 = 1024
    g1[2] = (int)(128u << 16);                      // tensor_dim1 = 128
    g1[3] = (int)(1024u << 16);                     // tile_dim0 = 1024
    g1[4] = 128;                                    // tile_dim1 = 128
    g1[5] = 1024;                                   // tensor_dim0_stride = 1024
    g1[6] = 0;
    g1[7] = 0;
    i32x4 z4 = (i32x4){0, 0, 0, 0};
#if __clang_major__ >= 23
    i32x8 z8 = (i32x8){0, 0, 0, 0, 0, 0, 0, 0};
    __builtin_amdgcn_tensor_load_to_lds(g0, g1, z4, z4, z8, 0);
#else
    __builtin_amdgcn_tensor_load_to_lds(g0, g1, z4, z4, 0);
#endif
    __builtin_amdgcn_s_wait_tensorcnt(0);
  }
#else
  // Fallback: per-lane async global->LDS copies (16B each), same padded image.
  {
    for (int i = 0; i < 64; ++i) {
      int c   = tid + i * 256;          // 16B chunk id, 16384 total
      int row = c >> 7;                 // 128 chunks per row
      int wb  = (c & 127) << 4;         // byte offset within row data
      unsigned ldsoff = (unsigned)(uintptr_t)lw + (unsigned)row * WROWB + wb +
                        (wb >= 1024 ? 8u : 0u);
      unsigned long long gaddr =
          (unsigned long long)(uintptr_t)W + (unsigned long long)row * 2048ull + wb;
      asm volatile("global_load_async_to_lds_b128 %0, %1, off"
                   :: "v"(ldsoff), "v"(gaddr) : "memory");
    }
    asm volatile("s_wait_asynccnt 0" ::: "memory");
  }
#endif
  __syncthreads();

  // ---- GEMM: 16(M) x 128(N) strip per wave; 2x-unrolled ping-pong ---------
  v8f acc[8];
#pragma unroll
  for (int n = 0; n < 8; ++n) acc[n] = (v8f){};

  const float* arow = X + (size_t)(m0 + r) * DM_;
  v16h a_cur = load_a_from_f32(arow + hi * 16);
  v16h bfA[8], bfB[8];
#pragma unroll
  for (int n = 0; n < 8; ++n)
    bfA[n] = load_b_contig(lds_w_ptr(lw, n * 16 + r, 0, hi));

  for (int k0 = 0; k0 < DM_; k0 += 64) {
    // -- stage 1: consume bfA(k0); prefetch bfB(k0+32) (always in range) --
    {
      const int nk = k0 + 32;
      v16h a_nx = load_a_from_f32(arow + nk + hi * 16);
#pragma unroll
      for (int n = 0; n < 8; ++n)
        bfB[n] = load_b_contig(lds_w_ptr(lw, n * 16 + r, nk, hi));
#pragma unroll
      for (int n = 0; n < 8; ++n)
        acc[n] = __builtin_amdgcn_wmma_f32_16x16x32_f16(
            false, a_cur, false, bfA[n], (short)0, acc[n], false, false);
      a_cur = a_nx;
#pragma unroll
      for (int n = 0; n < 8; ++n) { SGB(0x100, 2); SGB(0x008, 1); }
      SGB(0x020, 4);
    }
    // -- stage 2: consume bfB(k0+32); prefetch bfA(k0+64, clamped) --------
    {
      const int nk = (k0 + 64 < DM_) ? (k0 + 64) : 0;
      v16h a_nx = load_a_from_f32(arow + nk + hi * 16);
#pragma unroll
      for (int n = 0; n < 8; ++n)
        bfA[n] = load_b_contig(lds_w_ptr(lw, n * 16 + r, nk, hi));
#pragma unroll
      for (int n = 0; n < 8; ++n)
        acc[n] = __builtin_amdgcn_wmma_f32_16x16x32_f16(
            false, a_cur, false, bfB[n], (short)0, acc[n], false, false);
      a_cur = a_nx;
#pragma unroll
      for (int n = 0; n < 8; ++n) { SGB(0x100, 2); SGB(0x008, 1); }
      SGB(0x020, 4);
    }
  }

  if (sel < 2) {
    _Float16* O = (sel == 0) ? qf : kf;
#pragma unroll
    for (int n = 0; n < 8; ++n) {
      float bv = bia[n * 16 + r];
#pragma unroll
      for (int v = 0; v < 8; ++v) {
        int M = v + hi * 8;
        O[(size_t)(m0 + M) * DO_ + n * 16 + r] = (_Float16)(acc[n][v] + bv);
      }
    }
  } else {
    const int b  = m0 / SK_;
    const int kl = m0 % SK_;
#pragma unroll
    for (int n = 0; n < 8; ++n) {
      float bv = bia[n * 16 + r];
      v8h pack;
#pragma unroll
      for (int v = 0; v < 8; ++v) pack[v] = (_Float16)(acc[n][v] + bv);
      *(v8h*)(vt + ((size_t)b * DO_ + n * 16 + r) * SK_ + kl + hi * 8) = pack;
    }
  }
}

// ---------------------------------------------------------------------------
// K2: attention for one (batch, 16-query tile). 256 threads = 8 waves.
//     LDS: S[16][SROW] f32 + P[16][PROW] f16 + reductions (~198 KB).
// ---------------------------------------------------------------------------
static __device__ __forceinline__ void store_scores(
    float* S, v8f acc, int kidx, int hi, int vlen, float scale) {
  const bool masked = (kidx >= vlen);
#pragma unroll
  for (int v = 0; v < 8; ++v) {
    float s = acc[v] * scale;
    if (masked) s = MASKVAL;
    S[(v + hi * 8) * SROW + kidx] = s;
  }
}

__global__ __launch_bounds__(256)
void attn_kernel(const _Float16* __restrict__ qf,
                 const _Float16* __restrict__ kf,
                 const _Float16* __restrict__ vt,
                 const int* __restrict__ valid_lens,
                 float* __restrict__ out) {
  extern __shared__ unsigned char smem[];
  float*    S      = (float*)smem;                                  // 16*SROW f32
  _Float16* P      = (_Float16*)(smem + (size_t)16 * SROW * 4);     // 16*PROW f16
  float*    red    = (float*)(smem + (size_t)16 * SROW * 4 + (size_t)16 * PROW * 2);
  float*    rowmax = red + 256;
  float*    rowinv = rowmax + 16;

  const int tid  = threadIdx.x;
  const int lane = tid & 31;
  const int r    = lane & 15;
  const int hi   = lane >> 4;
  const int w    = tid >> 5;

  const int   b     = blockIdx.x >> 7;
  const int   q0    = (blockIdx.x & 127) * 16;
  const int   vlen  = valid_lens[b];
  const float scale = 0.088388347648318447f;  // 1/sqrt(128)

  const _Float16* Qb = qf + (size_t)b * SQ_ * DO_;
  const _Float16* Kb = kf + (size_t)b * SK_ * DO_;
  const _Float16* Vb = vt + (size_t)b * DO_ * SK_;

  v16h qfrag[4];
#pragma unroll
  for (int c = 0; c < 4; ++c)
    qfrag[c] = load_a_from_f16(Qb + (size_t)(q0 + r) * DO_ + c * 32 + hi * 16);

  // ---- Phase A: S = scale * Q K^T, masked; ping-pong over key tiles -------
  {
    v16h bfA[4], bfB[4];
#pragma unroll
    for (int c = 0; c < 4; ++c)
      bfA[c] = *(const v16h*)(Kb + (size_t)(w * 16 + r) * DO_ + c * 32 + hi * 16);

    for (int kt = w; kt < SK_ / 16; kt += 16) {  // 16 tiles/wave, 2 per trip
      // -- stage 1: consume bfA(kt); prefetch bfB(kt+8) (always in range) --
      {
        const int nkt = kt + 8;
        v8f acc = (v8f){};
#pragma unroll
        for (int c = 0; c < 4; ++c)
          bfB[c] = *(const v16h*)(Kb + (size_t)(nkt * 16 + r) * DO_ + c * 32 + hi * 16);
#pragma unroll
        for (int c = 0; c < 4; ++c)
          acc = __builtin_amdgcn_wmma_f32_16x16x32_f16(
              false, qfrag[c], false, bfA[c], (short)0, acc, false, false);
        store_scores(S, acc, kt * 16 + r, hi, vlen, scale);
        SGB(0x020, 8);
        SGB(0x008, 4);
      }
      // -- stage 2: consume bfB(kt+8); prefetch bfA(kt+16, clamped) --------
      {
        const int nkt = (kt + 16 < SK_ / 16) ? (kt + 16) : w;
        v8f acc = (v8f){};
#pragma unroll
        for (int c = 0; c < 4; ++c)
          bfA[c] = *(const v16h*)(Kb + (size_t)(nkt * 16 + r) * DO_ + c * 32 + hi * 16);
#pragma unroll
        for (int c = 0; c < 4; ++c)
          acc = __builtin_amdgcn_wmma_f32_16x16x32_f16(
              false, qfrag[c], false, bfB[c], (short)0, acc, false, false);
        store_scores(S, acc, (kt + 8) * 16 + r, hi, vlen, scale);
        SGB(0x020, 8);
        SGB(0x008, 4);
      }
    }
  }
  __syncthreads();

  // ---- Phase B: two-pass masked softmax (16 threads per row) --------------
  {
    const int row = tid >> 4;
    const int j   = tid & 15;
    float m = -3.0e38f;
    for (int i = 0; i < SK_ / 16; ++i)
      m = fmaxf(m, S[row * SROW + j + i * 16]);
    red[row * 16 + j] = m;
    __syncthreads();
    if (j == 0) {
      float mm = red[row * 16];
      for (int t = 1; t < 16; ++t) mm = fmaxf(mm, red[row * 16 + t]);
      rowmax[row] = mm;
    }
    __syncthreads();
    const float mrow = rowmax[row];
    float ssum = 0.0f;
    for (int i = 0; i < SK_ / 16; ++i) {
      const int col = j + i * 16;
      float e = __expf(S[row * SROW + col] - mrow);
      P[row * PROW + col] = (_Float16)e;
      ssum += e;
    }
    red[row * 16 + j] = ssum;
    __syncthreads();
    if (j == 0) {
      float t = 0.0f;
      for (int q = 0; q < 16; ++q) t += red[row * 16 + q];
      rowinv[row] = 1.0f / t;
    }
    __syncthreads();
  }

  // ---- Phase C: out = (P V) * rowinv; wave w owns d-cols [16w,16w+16) -----
  {
    const int n0 = w * 16;
    v8f acc = (v8f){};
    v16h aA = load_a_from_f16(P + r * PROW + hi * 16);
    v16h bA = *(const v16h*)(Vb + (size_t)(n0 + r) * SK_ + hi * 16);
    v16h aB, bB;
    for (int kk = 0; kk < SK_; kk += 64) {  // 2 k-steps per trip, ping-pong
      // -- stage 1: consume (aA,bA); prefetch kk+32 (always in range) ------
      {
        const int nk = kk + 32;
        aB = load_a_from_f16(P + r * PROW + nk + hi * 16);
        bB = *(const v16h*)(Vb + (size_t)(n0 + r) * SK_ + nk + hi * 16);
        acc = __builtin_amdgcn_wmma_f32_16x16x32_f16(
            false, aA, false, bA, (short)0, acc, false, false);
        SGB(0x100, 2); SGB(0x020, 2); SGB(0x008, 1);
      }
      // -- stage 2: consume (aB,bB); prefetch kk+64 (clamped) --------------
      {
        const int nk = (kk + 64 < SK_) ? (kk + 64) : 0;
        aA = load_a_from_f16(P + r * PROW + nk + hi * 16);
        bA = *(const v16h*)(Vb + (size_t)(n0 + r) * SK_ + nk + hi * 16);
        acc = __builtin_amdgcn_wmma_f32_16x16x32_f16(
            false, aB, false, bB, (short)0, acc, false, false);
        SGB(0x100, 2); SGB(0x020, 2); SGB(0x008, 1);
      }
    }
#pragma unroll
    for (int v = 0; v < 8; ++v) {
      const int M = v + hi * 8;
      out[((size_t)b * SQ_ + q0 + M) * DO_ + n0 + r] = acc[v] * rowinv[M];
    }
  }
}

// ---------------------------------------------------------------------------
extern "C" void kernel_launch(void* const* d_in, const int* in_sizes, int n_in,
                              void* d_out, int out_size, void* d_ws, size_t ws_size,
                              hipStream_t stream) {
  (void)in_sizes; (void)n_in; (void)out_size; (void)ws_size;

  const float* queries = (const float*)d_in[0];
  const float* keys    = (const float*)d_in[1];
  const float* values  = (const float*)d_in[2];
  const int*   vlens   = (const int*)d_in[3];
  const float* w_q     = (const float*)d_in[4];
  const float* b_q     = (const float*)d_in[5];
  const float* w_k     = (const float*)d_in[6];
  const float* b_k     = (const float*)d_in[7];
  const float* w_v     = (const float*)d_in[8];
  const float* b_v     = (const float*)d_in[9];
  float*       out     = (float*)d_out;

  // Workspace (f16): Wt[3][128][1024] | Q[B*SQ][128] | K[B*SK][128] | Vt[B][128][SK]
  _Float16* wt = (_Float16*)d_ws;
  _Float16* qf = wt + (size_t)3 * DM_ * DO_;
  _Float16* kf = qf + (size_t)B_ * SQ_ * DO_;
  _Float16* vt = kf + (size_t)B_ * SK_ * DO_;

  wconv_kernel<<<(3 * DM_ * DO_ + 255) / 256, 256, 0, stream>>>(w_q, w_k, w_v, wt);

  const size_t shmem1 = (size_t)DO_ * WROWB;  // 264192 B padded weight image
  qkv_proj_kernel<<<(3 * 1024) / 8, 256, shmem1, stream>>>(
      queries, keys, values, b_q, b_k, b_v, wt, qf, kf, vt);

  const size_t shmem2 = (size_t)16 * SROW * 4 + (size_t)16 * PROW * 2 + (256 + 32) * 4;
  attn_kernel<<<B_ * (SQ_ / 16), 256, shmem2, stream>>>(qf, kf, vt, vlens, out);
}
